// LshSelfAttention_10093173145851
// MI455X (gfx1250) — matrix operand order repaired and checked by reading, not compiled, not run on hardware
//
#include <hip/hip_runtime.h>

typedef __attribute__((ext_vector_type(16))) __bf16 v16bf;
typedef __attribute__((ext_vector_type(8)))  float  v8f;
typedef __attribute__((ext_vector_type(4)))  float  f32x4;
typedef __attribute__((ext_vector_type(4)))  unsigned int u32x4;
typedef __attribute__((ext_vector_type(2)))  unsigned int u32x2;

#define DEVINL __device__ __forceinline__

constexpr int TT     = 4224;            // padded sequence length
constexpr int LL     = 4096;            // valid sequence length
constexpr int DMODEL = 512;
constexpr int NHEADS = 8;
constexpr int DH     = 64;
constexpr int BATCH  = 4;
constexpr int BB     = BATCH * NHEADS;  // 32
constexpr int NHASH  = 4;
constexpr int NBUCK  = 66;              // buckets per hash round
constexpr int NBTOT  = NHASH * NBUCK;   // 264
constexpr int NHT    = NHASH * TT;      // 16896 sorted slots per B
constexpr int NCHUNK = NHT / 64;        // 264 chunks per B

DEVINL unsigned short f2bf(float f) {
  unsigned int u = __float_as_uint(f);
  u += 0x7fffu + ((u >> 16) & 1u);      // round-to-nearest-even
  return (unsigned short)(u >> 16);
}
DEVINL unsigned int pack2(float a, float b) {
  return (unsigned int)f2bf(a) | ((unsigned int)f2bf(b) << 16);
}

// ---- CDNA5 async global->LDS (ASYNCcnt) -----------------------------------
// LDS destination VGPR holds the 32-bit LDS byte offset (added to the
// LDS base and the instruction offset; flat shared addr low 32 bits == LDS offset).
DEVINL void async_ld16(unsigned ldsoff, const void* g) {
  asm volatile("global_load_async_to_lds_b128 %0, %1, off"
               :: "v"(ldsoff), "v"((unsigned long long)(size_t)g)
               : "memory");
}
#define WAIT_ASYNC(n) asm volatile("s_wait_asynccnt " #n ::: "memory")

// A-fragment (16x32 bf16): lane L -> row L%16; elems 0..7 = K base+{0|8}..,
// elems 8..15 = K base+{16|24}.. depending on lane half.
DEVINL v16bf frag_a(const unsigned short* p, int row, int ld, int kk, int lane) {
  const int k0 = kk + ((lane & 16) ? 8 : 0);
  union { u32x4 q[2]; v16bf v; } f;
  f.q[0] = *(const u32x4*)(p + row * ld + k0);
  f.q[1] = *(const u32x4*)(p + row * ld + k0 + 16);
  return f.v;
}
// B-fragment (32x16 bf16): lane L -> col L%16; elems 0..7 = K base+{0|8}..,
// elems 8..15 = K base+{16|24}.. depending on lane half.
DEVINL v16bf frag_b(const unsigned short* p, int row, int ld, int kk, int lane) {
  const int k0 = kk + ((lane & 16) ? 8 : 0);
  union { u32x4 q[2]; v16bf v; } f;
  f.q[0] = *(const u32x4*)(p + row * ld + k0);
  f.q[1] = *(const u32x4*)(p + row * ld + k0 + 16);
  return f.v;
}
DEVINL v8f wmma_bf16(v16bf a, v16bf b, v8f c) {
  return __builtin_amdgcn_wmma_f32_16x16x32_bf16(false, a, false, b, (short)0, c,
                                                 false, false);
}

// ---------------------------------------------------------------------------
// Kernel 0: f32 -> bf16 conversion (x and the three weight matrices).
// ---------------------------------------------------------------------------
__global__ __launch_bounds__(256) void convert_kernel(
    const float* __restrict__ src, unsigned short* __restrict__ dst, int n4)
{
  int i = blockIdx.x * 256 + threadIdx.x;
  const int stride = gridDim.x * 256;
  for (; i < n4; i += stride) {
    f32x4 v4 = *(const f32x4*)(src + (size_t)i * 4);
    u32x2 p; p.x = pack2(v4.x, v4.y); p.y = pack2(v4.z, v4.w);
    *(u32x2*)(dst + (size_t)i * 4) = p;
  }
}

// ---------------------------------------------------------------------------
// Kernel 1: fused QK / V projection (NT GEMM, bf16 in, f32 out head-split).
// Double-buffered LDS tiles streamed with global_load_async_to_lds_b128.
// grid = (33, 4, 8)  z = bi*2 + which(0=qk,1=v);  256 threads = 8 waves.
// ---------------------------------------------------------------------------
__global__ __launch_bounds__(256) void proj_kernel(
    const unsigned short* __restrict__ xbf, const unsigned short* __restrict__ wqk,
    const unsigned short* __restrict__ wv, float* __restrict__ qk,
    float* __restrict__ vout)
{
  __shared__ unsigned short As[2 * 128 * 40];   // 2 buffers, row stride 40
  __shared__ unsigned short Bs[2 * 128 * 40];
  const int m0 = blockIdx.x * 128;
  const int e0 = blockIdx.y * 128;
  const int bi = blockIdx.z >> 1;
  const int which = blockIdx.z & 1;
  const unsigned short* __restrict__ W = which ? wv : wqk;
  float* __restrict__ O = which ? vout : qk;
  const int tid = threadIdx.x, lane = tid & 31, wave = tid >> 5;
  const int pm = wave & 3, nh = wave >> 2;
  const int hi = (lane >> 4) & 1;
  const bool padA = (m0 >= LL);                 // last M tile: all zero padding

  const unsigned ldsA = (unsigned)(size_t)(void*)As;
  const unsigned ldsB = (unsigned)(size_t)(void*)Bs;

  if (padA) {
    for (int i = tid; i < 2 * 128 * 40; i += 256) As[i] = 0;
    __syncthreads();
  }

  const int rrow = tid >> 2, rc16 = tid & 3;    // 16B chunk mapping
  auto issue = [&](int buf, int k0) {
#pragma unroll
    for (int i = 0; i < 2; ++i) {
      const int row = rrow + i * 64;
      const unsigned dsts = (unsigned)(buf * 10240 + row * 80 + rc16 * 16);
      if (!padA)
        async_ld16(ldsA + dsts,
                   xbf + ((size_t)bi * LL + m0 + row) * DMODEL + k0 + rc16 * 8);
      async_ld16(ldsB + dsts, W + (size_t)(e0 + row) * DMODEL + k0 + rc16 * 8);
    }
  };

  const v8f vzero = {0.f,0.f,0.f,0.f,0.f,0.f,0.f,0.f};
  v8f acc[2][4];
#pragma unroll
  for (int mi = 0; mi < 2; ++mi)
#pragma unroll
    for (int ni = 0; ni < 4; ++ni) acc[mi][ni] = vzero;

  issue(0, 0);
  for (int it = 0; it < DMODEL / 32; ++it) {
    if (it + 1 < DMODEL / 32) {
      issue((it + 1) & 1, (it + 1) * 32);
      if (padA) { WAIT_ASYNC(2); } else { WAIT_ASYNC(4); }
    } else {
      WAIT_ASYNC(0);
    }
    __syncthreads();
    const unsigned short* Ab = As + (it & 1) * 5120;
    const unsigned short* Bb = Bs + (it & 1) * 5120;
    v16bf a0 = frag_a(Ab, 32 * pm + (lane & 15), 40, 0, lane);
    v16bf a1 = frag_a(Ab, 32 * pm + 16 + (lane & 15), 40, 0, lane);
#pragma unroll
    for (int ni = 0; ni < 4; ++ni) {
      v16bf b = frag_b(Bb, 64 * nh + 16 * ni + (lane & 15), 40, 0, lane);
      acc[0][ni] = wmma_bf16(a0, b, acc[0][ni]);
      acc[1][ni] = wmma_bf16(a1, b, acc[1][ni]);
    }
    __syncthreads();
  }
#pragma unroll
  for (int mi = 0; mi < 2; ++mi)
#pragma unroll
    for (int ni = 0; ni < 4; ++ni) {
      const int e = e0 + 64 * nh + 16 * ni + (lane & 15);
      const int head = e >> 6, dc = e & 63;
#pragma unroll
      for (int r = 0; r < 8; ++r) {
        const int m = m0 + 32 * pm + 16 * mi + r + 8 * hi;
        O[((size_t)(bi * NHEADS + head) * TT + m) * DH + dc] = acc[mi][ni][r];
      }
    }
}

// ---------------------------------------------------------------------------
// Kernel 2: LSH hashing.  rot = qk . rotations; bucket = argmax([rot,-rot]) + h*66
// ---------------------------------------------------------------------------
__global__ __launch_bounds__(256) void hash_kernel(
    const float* __restrict__ qk, const float* __restrict__ rot,
    int* __restrict__ bucket)
{
  __shared__ float R[DH * NHASH * 33];           // 64*132 f32 = 33 KB
  for (int i = threadIdx.x; i < DH * NHASH * 33; i += 256) R[i] = rot[i];
  __syncthreads();
  const int gid = blockIdx.x * 256 + threadIdx.x;   // 528*256 == 32*4224 exactly
  const int Bi = gid / TT, t = gid % TT;
  const float* src = qk + ((size_t)Bi * TT + t) * DH;
  f32x4 qv[16];
#pragma unroll
  for (int i = 0; i < 16; ++i) qv[i] = *(const f32x4*)(src + i * 4);
  for (int h = 0; h < NHASH; ++h) {
    float bp = -3.0e38f, bn = -3.0e38f;
    int ip = 0, in_ = 0;
    for (int i = 0; i < 33; ++i) {
      float a = 0.f;
#pragma unroll
      for (int fq = 0; fq < 16; ++fq) {
        const float* rp = &R[(fq * 4) * 132 + h * 33 + i];
        a = fmaf(qv[fq].x, rp[0],   a);
        a = fmaf(qv[fq].y, rp[132], a);
        a = fmaf(qv[fq].z, rp[264], a);
        a = fmaf(qv[fq].w, rp[396], a);
      }
      if (a > bp)  { bp = a;  ip  = i; }
      if (-a > bn) { bn = -a; in_ = i; }
    }
    const int bidx = (bp >= bn) ? ip : (33 + in_);  // concat([rot,-rot]) argmax
    bucket[(size_t)Bi * NHT + h * TT + t] = h * NBUCK + bidx;
  }
}

// ---------------------------------------------------------------------------
// Kernel 3: stable counting sort by bucket (keys unique, within-bucket order = t)
// ---------------------------------------------------------------------------
__global__ __launch_bounds__(256) void sort_kernel(
    const int* __restrict__ bucket, int* __restrict__ st, int* __restrict__ undo)
{
  __shared__ unsigned short lb[NHT];   // 33.8 KB
  __shared__ int hist[NBTOT];
  __shared__ int offs[NBTOT];
  const int Bi = blockIdx.x, tid = threadIdx.x;
  const size_t base = (size_t)Bi * NHT;
  for (int i = tid; i < NBTOT; i += 256) hist[i] = 0;
  __syncthreads();
  for (int i = tid; i < NHT; i += 256) {
    const int b = bucket[base + i];
    lb[i] = (unsigned short)b;
    atomicAdd(&hist[b], 1);
  }
  __syncthreads();
  if (tid == 0) {
    int run = 0;
    for (int j = 0; j < NBTOT; ++j) { offs[j] = run; run += hist[j]; }
  }
  __syncthreads();
  for (int b = tid; b < NBTOT; b += 256) {
    int pos = offs[b];
    int idx = 0;
    for (int h = 0; h < NHASH; ++h)
      for (int t = 0; t < TT; ++t, ++idx)
        if ((int)lb[idx] == b) {
          st[base + pos]   = t;
          undo[base + idx] = pos;
          ++pos;
        }
  }
}

// ---------------------------------------------------------------------------
// Kernel 4: per-chunk attention.  grid = (264 chunks, 32 B); 256 threads.
// ---------------------------------------------------------------------------
__global__ __launch_bounds__(256) void attn_kernel(
    const float* __restrict__ qk, const float* __restrict__ vin,
    const int* __restrict__ st, float* __restrict__ bo,
    float* __restrict__ lse_out)
{
  __shared__ unsigned short QS[64 * 72];
  __shared__ unsigned short KS[128 * 72];
  __shared__ unsigned short VT[64 * 136];   // V transposed: [dcol][key j]
  __shared__ unsigned short PS[64 * 136];   // probs bf16
  __shared__ float rowmax2[2][64];
  __shared__ float rowsum2[2][64];
  __shared__ int   tqs[64];
  __shared__ int   tks[128];

  const int c = blockIdx.x, Bi = blockIdx.y;
  const int tid = threadIdx.x, lane = tid & 31, wave = tid >> 5;
  const int hi = (lane >> 4) & 1;
  const size_t baseS = (size_t)Bi * NHT;

  if (tid < 64) tqs[tid] = st[baseS + c * 64 + tid];
  if (tid >= 64 && tid < 192) {
    const int j = tid - 64;
    const int cj = (j < 64) ? c : (c + NCHUNK - 1) % NCHUNK;  // look-back chunk
    tks[j] = st[baseS + cj * 64 + (j & 63)];
  }
  __syncthreads();

  { // Q gather, scaled by Dh^-0.5
    const int row = tid >> 2, part = tid & 3;
    const float* src = qk + ((size_t)Bi * TT + tqs[row]) * DH + part * 16;
    unsigned short* dst = &QS[row * 72 + part * 16];
#pragma unroll
    for (int i = 0; i < 4; ++i) {
      f32x4 v4 = *(const f32x4*)(src + i * 4);
      u32x2 p; p.x = pack2(v4.x * 0.125f, v4.y * 0.125f);
      p.y = pack2(v4.z * 0.125f, v4.w * 0.125f);
      *(u32x2*)(dst + i * 4) = p;
    }
  }
  { // K gather + L2-normalize rows (pairs of threads share one key row)
    const int row = tid >> 1, half = tid & 1;
    const float* src = qk + ((size_t)Bi * TT + tks[row]) * DH + half * 32;
    float ss = 0.f;
#pragma unroll
    for (int i = 0; i < 8; ++i) {
      f32x4 v4 = *(const f32x4*)(src + i * 4);
      ss = fmaf(v4.x, v4.x, fmaf(v4.y, v4.y, fmaf(v4.z, v4.z, fmaf(v4.w, v4.w, ss))));
    }
    ss += __shfl_xor(ss, 1, 32);
    const float sc = 1.0f / fmaxf(sqrtf(ss), 1e-12f);
    unsigned short* dst = &KS[row * 72 + half * 32];
#pragma unroll
    for (int i = 0; i < 8; ++i) {
      f32x4 v4 = *(const f32x4*)(src + i * 4);
      u32x2 p; p.x = pack2(v4.x * sc, v4.y * sc); p.y = pack2(v4.z * sc, v4.w * sc);
      *(u32x2*)(dst + i * 4) = p;
    }
  }
  { // V gather, transposed into VT[dcol][j]
    const int row = tid >> 1, half = tid & 1;
    const float* src = vin + ((size_t)Bi * TT + tks[row]) * DH + half * 32;
#pragma unroll
    for (int i = 0; i < 8; ++i) {
      f32x4 v4 = *(const f32x4*)(src + i * 4);
      VT[(half * 32 + i * 4 + 0) * 136 + row] = f2bf(v4.x);
      VT[(half * 32 + i * 4 + 1) * 136 + row] = f2bf(v4.y);
      VT[(half * 32 + i * 4 + 2) * 136 + row] = f2bf(v4.z);
      VT[(half * 32 + i * 4 + 3) * 136 + row] = f2bf(v4.w);
    }
  }
  __syncthreads();

  // GEMM1: S = Q x K^T  (wave -> 16-row block rb, 64-col half ch)
  const int rb = wave & 3, ch = wave >> 2;
  const v8f vzero = {0.f,0.f,0.f,0.f,0.f,0.f,0.f,0.f};
  v8f sacc[4];
#pragma unroll
  for (int ni = 0; ni < 4; ++ni) sacc[ni] = vzero;
#pragma unroll
  for (int kk = 0; kk < 64; kk += 32) {
    v16bf a = frag_a(QS, 16 * rb + (lane & 15), 72, kk, lane);
#pragma unroll
    for (int ni = 0; ni < 4; ++ni) {
      v16bf b = frag_b(KS, 64 * ch + 16 * ni + (lane & 15), 72, kk, lane);
      sacc[ni] = wmma_bf16(a, b, sacc[ni]);
    }
  }
  // self-token mask + per-half row max
  int tkn[4];
#pragma unroll
  for (int ni = 0; ni < 4; ++ni) tkn[ni] = tks[64 * ch + 16 * ni + (lane & 15)];
#pragma unroll
  for (int r = 0; r < 8; ++r) {
    const int tqm = tqs[16 * rb + r + 8 * hi];
    float mx = -3.0e38f;
#pragma unroll
    for (int ni = 0; ni < 4; ++ni) {
      float val = (tqm == tkn[ni]) ? -5.0e4f : sacc[ni][r];
      sacc[ni][r] = val;
      mx = fmaxf(mx, val);
    }
    for (int off = 1; off < 16; off <<= 1) mx = fmaxf(mx, __shfl_xor(mx, off, 32));
    if ((lane & 15) == 0) rowmax2[ch][16 * rb + r + 8 * hi] = mx;
  }
  __syncthreads();
  // exp(s - gmax), probs -> PS (bf16), per-half row sums
#pragma unroll
  for (int r = 0; r < 8; ++r) {
    const int m = 16 * rb + r + 8 * hi;
    const float gm = fmaxf(rowmax2[0][m], rowmax2[1][m]);
    float sum = 0.f;
#pragma unroll
    for (int ni = 0; ni < 4; ++ni) {
      const float p = __expf(sacc[ni][r] - gm);
      PS[m * 136 + 64 * ch + 16 * ni + (lane & 15)] = f2bf(p);
      sum += p;
    }
    for (int off = 1; off < 16; off <<= 1) sum += __shfl_xor(sum, off, 32);
    if ((lane & 15) == 0) rowsum2[ch][m] = sum;
  }
  __syncthreads();
  if (tid < 64) {
    const float gm = fmaxf(rowmax2[0][tid], rowmax2[1][tid]);
    const float gs = rowsum2[0][tid] + rowsum2[1][tid];
    lse_out[baseS + c * 64 + tid] = gm + __logf(gs);
  }
  // GEMM2: O = P x V  (wave -> 16-row block mt, 32-col half nh2)
  const int mt = wave & 3, nh2 = wave >> 2;
  v8f oacc[2];
  oacc[0] = vzero; oacc[1] = vzero;
#pragma unroll
  for (int kk = 0; kk < 128; kk += 32) {
    v16bf a = frag_a(PS, 16 * mt + (lane & 15), 136, kk, lane);
#pragma unroll
    for (int ni = 0; ni < 2; ++ni) {
      v16bf b = frag_b(VT, 32 * nh2 + 16 * ni + (lane & 15), 136, kk, lane);
      oacc[ni] = wmma_bf16(a, b, oacc[ni]);
    }
  }
#pragma unroll
  for (int ni = 0; ni < 2; ++ni)
#pragma unroll
    for (int r = 0; r < 8; ++r) {
      const int m = 16 * mt + r + 8 * hi;
      const float inv = 1.0f / (rowsum2[0][m] + rowsum2[1][m]);
      const int ncol = 32 * nh2 + 16 * ni + (lane & 15);
      bo[(baseS + c * 64 + m) * DH + ncol] = oacc[ni][r] * inv;
    }
}

// ---------------------------------------------------------------------------
// Kernel 5: unsort + combine hash rounds; writes merged-head context in bf16.
// 32 threads per (Bi, t<4096) row, 2 elements per thread.
// ---------------------------------------------------------------------------
__global__ __launch_bounds__(256) void combine_kernel(
    const float* __restrict__ bo, const float* __restrict__ lse,
    const int* __restrict__ undo, unsigned short* __restrict__ comb)
{
  const int r  = blockIdx.x * 8 + (threadIdx.x >> 5);
  const int d0 = (threadIdx.x & 31) * 2;
  const int Bi = r >> 12;              // /4096
  const int t  = r & 4095;
  const size_t base = (size_t)Bi * NHT;
  int pos[NHASH]; float ls[NHASH];
#pragma unroll
  for (int h = 0; h < NHASH; ++h) {
    pos[h] = undo[base + h * TT + t];
    ls[h]  = lse[base + pos[h]];
  }
  const float m = fmaxf(fmaxf(ls[0], ls[1]), fmaxf(ls[2], ls[3]));
  float e[NHASH], se = 0.f;
#pragma unroll
  for (int h = 0; h < NHASH; ++h) { e[h] = __expf(ls[h] - m); se += e[h]; }
  float a0 = 0.f, a1 = 0.f;
#pragma unroll
  for (int h = 0; h < NHASH; ++h) {
    const float w = e[h] / se;
    const float* op = bo + (base + pos[h]) * DH + d0;
    a0 += w * op[0];
    a1 += w * op[1];
  }
  const int head = Bi & 7, bi = Bi >> 3;
  *(unsigned int*)(comb + ((size_t)bi * LL + t) * DMODEL + head * DH + d0) =
      pack2(a0, a1);
}

// ---------------------------------------------------------------------------
// Kernel 6: output projection + bias (bf16 in, f32 out), t<4096 rows only.
// Same async double-buffered pipeline as proj_kernel; grid = (32, 4, 4).
// ---------------------------------------------------------------------------
__global__ __launch_bounds__(256) void outproj_kernel(
    const unsigned short* __restrict__ comb, const unsigned short* __restrict__ w_out,
    const float* __restrict__ b_out, float* __restrict__ out)
{
  __shared__ unsigned short As[2 * 128 * 40];
  __shared__ unsigned short Bs[2 * 128 * 40];
  const int m0 = blockIdx.x * 128;
  const int e0 = blockIdx.y * 128;
  const int bi = blockIdx.z;
  const int tid = threadIdx.x, lane = tid & 31, wave = tid >> 5;
  const int pm = wave & 3, nh = wave >> 2;
  const int hi = (lane >> 4) & 1;

  const unsigned ldsA = (unsigned)(size_t)(void*)As;
  const unsigned ldsB = (unsigned)(size_t)(void*)Bs;

  const int rrow = tid >> 2, rc16 = tid & 3;
  auto issue = [&](int buf, int k0) {
#pragma unroll
    for (int i = 0; i < 2; ++i) {
      const int row = rrow + i * 64;
      const unsigned dsts = (unsigned)(buf * 10240 + row * 80 + rc16 * 16);
      async_ld16(ldsA + dsts,
                 comb + ((size_t)bi * LL + m0 + row) * DMODEL + k0 + rc16 * 8);
      async_ld16(ldsB + dsts, w_out + (size_t)(e0 + row) * DMODEL + k0 + rc16 * 8);
    }
  };

  const v8f vzero = {0.f,0.f,0.f,0.f,0.f,0.f,0.f,0.f};
  v8f acc[2][4];
#pragma unroll
  for (int mi = 0; mi < 2; ++mi)
#pragma unroll
    for (int ni = 0; ni < 4; ++ni) acc[mi][ni] = vzero;

  issue(0, 0);
  for (int it = 0; it < DMODEL / 32; ++it) {
    if (it + 1 < DMODEL / 32) {
      issue((it + 1) & 1, (it + 1) * 32);
      WAIT_ASYNC(4);
    } else {
      WAIT_ASYNC(0);
    }
    __syncthreads();
    const unsigned short* Ab = As + (it & 1) * 5120;
    const unsigned short* Bb = Bs + (it & 1) * 5120;
    v16bf a0 = frag_a(Ab, 32 * pm + (lane & 15), 40, 0, lane);
    v16bf a1 = frag_a(Ab, 32 * pm + 16 + (lane & 15), 40, 0, lane);
#pragma unroll
    for (int ni = 0; ni < 4; ++ni) {
      v16bf b = frag_b(Bb, 64 * nh + 16 * ni + (lane & 15), 40, 0, lane);
      acc[0][ni] = wmma_bf16(a0, b, acc[0][ni]);
      acc[1][ni] = wmma_bf16(a1, b, acc[1][ni]);
    }
    __syncthreads();
  }
#pragma unroll
  for (int mi = 0; mi < 2; ++mi)
#pragma unroll
    for (int ni = 0; ni < 4; ++ni) {
      const int e = e0 + 64 * nh + 16 * ni + (lane & 15);
      const float bias = b_out[e];
#pragma unroll
      for (int r = 0; r < 8; ++r) {
        const int m = m0 + 32 * pm + 16 * mi + r + 8 * hi;
        out[((size_t)bi * LL + m) * DMODEL + e] = acc[mi][ni][r] + bias;
      }
    }
}

// ---------------------------------------------------------------------------
extern "C" void kernel_launch(void* const* d_in, const int* in_sizes, int n_in,
                              void* d_out, int out_size, void* d_ws, size_t ws_size,
                              hipStream_t stream)
{
  const float* x     = (const float*)d_in[0];
  const float* rotw  = (const float*)d_in[1];
  const float* w_qk  = (const float*)d_in[2];
  const float* w_v   = (const float*)d_in[3];
  const float* w_out = (const float*)d_in[4];
  const float* bias  = (const float*)d_in[5];
  float* out = (float*)d_out;

  char* ws = (char*)d_ws;
  size_t off = 0;
  auto carve = [&](size_t bytes) -> void* {
    void* p = ws + off;
    off += (bytes + 255) & ~(size_t)255;
    return p;
  };
  float* qk   = (float*)carve(sizeof(float) * (size_t)BB * TT * DH);
  float* vbuf = (float*)carve(sizeof(float) * (size_t)BB * TT * DH);
  int*   bkt  = (int*)  carve(sizeof(int)   * (size_t)BB * NHT);
  int*   st   = (int*)  carve(sizeof(int)   * (size_t)BB * NHT);
  int*   undo = (int*)  carve(sizeof(int)   * (size_t)BB * NHT);
  float* bo   = (float*)carve(sizeof(float) * (size_t)BB * NHT * DH);
  float* lse  = (float*)carve(sizeof(float) * (size_t)BB * NHT);
  unsigned short* xbf  = (unsigned short*)carve(2u * (size_t)BATCH * LL * DMODEL);
  unsigned short* wqkb = (unsigned short*)carve(2u * (size_t)DMODEL * DMODEL);
  unsigned short* wvb  = (unsigned short*)carve(2u * (size_t)DMODEL * DMODEL);
  unsigned short* wob  = (unsigned short*)carve(2u * (size_t)DMODEL * DMODEL);
  unsigned short* cmbb = (unsigned short*)carve(2u * (size_t)BATCH * LL * DMODEL);

  convert_kernel<<<dim3(2048), 256, 0, stream>>>(x, xbf, BATCH * LL * DMODEL / 4);
  convert_kernel<<<dim3(256), 256, 0, stream>>>(w_qk, wqkb, DMODEL * DMODEL / 4);
  convert_kernel<<<dim3(256), 256, 0, stream>>>(w_v, wvb, DMODEL * DMODEL / 4);
  convert_kernel<<<dim3(256), 256, 0, stream>>>(w_out, wob, DMODEL * DMODEL / 4);

  proj_kernel   <<<dim3(TT / 128, DMODEL / 128, BATCH * 2), 256, 0, stream>>>(xbf, wqkb, wvb, qk, vbuf);
  hash_kernel   <<<dim3((BB * TT) / 256), 256, 0, stream>>>(qk, rotw, bkt);
  sort_kernel   <<<dim3(BB), 256, 0, stream>>>(bkt, st, undo);
  attn_kernel   <<<dim3(NCHUNK, BB), 256, 0, stream>>>(qk, vbuf, st, bo, lse);
  combine_kernel<<<dim3((BB * LL) / 8), 256, 0, stream>>>(bo, lse, undo, cmbb);
  outproj_kernel<<<dim3(LL / 128, DMODEL / 128, BATCH), 256, 0, stream>>>(cmbb, wob, bias, out);

  (void)in_sizes; (void)n_in; (void)out_size; (void)ws_size;
}
